// Attention_40664750358685
// MI455X (gfx1250) — matrix operand order repaired and checked by reading, not compiled, not run on hardware
//
#include <hip/hip_runtime.h>

// ---------------------------------------------------------------------------
// Attention block for MI455X (gfx1250, wave32, WMMA bf16 -> f32 accumulate)
//   B=2, T=2048, D=1024, H=16, hd=64
// Pipeline:
//   1) convert x -> bf16; convert+transpose wq/wk/wv/wo -> bf16 [N,K]
//   2) QKV GEMM (WMMA 16x16x32 bf16, 2 M-tiles/wave), fused per-head RMSNorm
//      epilogue, Q scaled by 1/8; V written transposed [B,H,64,T]
//   3) flash attention: K/V key-blocks staged to LDS via TDM
//      (tensor_load_to_lds, double-buffered, s_wait_tensorcnt + barrier),
//      S^T = K_tile x Q^T (WMMA), online softmax, O += P x V (WMMA; the S^T
//      C-layout maps onto the P A-layout with zero cross-lane moves)
//   4) output GEMM (WMMA, 2 M-tiles/wave) -> f32 d_out
// ---------------------------------------------------------------------------

typedef __attribute__((ext_vector_type(16))) __bf16 v16bf;
typedef __attribute__((ext_vector_type(8)))  __bf16 v8bf;
typedef __attribute__((ext_vector_type(8)))  float  v8f;
typedef __attribute__((ext_vector_type(4)))  unsigned int v4u;
typedef __attribute__((ext_vector_type(8)))  int v8i;
typedef __attribute__((ext_vector_type(4)))  int v4i;

#define BATCH 2
#define SEQ   2048
#define DMODEL 1024
#define NHEADS 16
#define HDIM  64
#define MROWS (BATCH*SEQ)   // 4096

// float -> bf16 (round to nearest even, bits form)
__device__ __forceinline__ unsigned short f2bfu(float f) {
    union { float f; unsigned u; } in; in.f = f;
    unsigned u = in.u;
    unsigned r = u + 0x7FFFu + ((u >> 16) & 1u);
    return (unsigned short)(r >> 16);
}

__device__ __forceinline__ __bf16 f2bf(float f) {
    union { unsigned short s; __bf16 b; } o; o.s = f2bfu(f); return o.b;
}

__device__ __forceinline__ v8f wmma_bf16(v16bf a, v16bf b, v8f c) {
    // (neg_a, A, neg_b, B, c_mod, C, reuse_a, reuse_b)
    return __builtin_amdgcn_wmma_f32_16x16x32_bf16(false, a, false, b,
                                                   (short)0, c, false, false);
}

// Build a 16-bit A-operand (16x32 slice) from two contiguous 8-elem chunks:
// p = row_base + k0 + 8*laneHalf; chunks cover k = off..off+7 and
// 16+off..16+off+7 per the ISA A-matrix layout.
__device__ __forceinline__ v16bf make_a(const __bf16* p) {
    v8bf lo = *(const v8bf*)(p);
    v8bf hi = *(const v8bf*)(p + 16);
    v16bf a;
#pragma unroll
    for (int j = 0; j < 8; ++j) { a[j] = lo[j]; a[8 + j] = hi[j]; }
    return a;
}

// ---------------------------------------------------------------------------
// TDM: 2D tile load Global -> LDS (D# per CDNA5 ISA ch.8; 2-byte elements).
// Groups 2/3 zero: 2D tensor, tile_dim2=0 => unused. workgroup_mask=0 (no
// cluster). count=1, type=2 ("image"), data_size code 1 (2 bytes).
// This toolchain exposes the 6-arg builtin:
//   (uint32x4 g0, int32x8 g1, int32x4, int32x4, int32x8, i32 cpol)
// ---------------------------------------------------------------------------
__device__ __forceinline__ void tdm_load_2d(unsigned lds_off, const void* gaddr,
                                            unsigned tensor_d0, unsigned tensor_d1,
                                            unsigned tile_d0, unsigned tile_d1,
                                            unsigned stride0 /* elements */) {
    unsigned long long ga = (unsigned long long)gaddr;
    v4u g0;
    g0[0] = 1u;                                            // count=1
    g0[1] = lds_off;                                       // lds_addr (bytes)
    g0[2] = (unsigned)(ga & 0xFFFFFFFFu);                  // global_addr lo
    g0[3] = (unsigned)((ga >> 32) & 0x01FFFFFFu) | (2u << 30); // addr hi | type=2
    v8i g1;
    g1[0] = (int)(1u << 16);                               // data_size=2B, mask=0
    g1[1] = (int)((tensor_d0 & 0xFFFFu) << 16);            // tensor_dim0 lo16
    g1[2] = (int)((tensor_d0 >> 16) | ((tensor_d1 & 0xFFFFu) << 16));
    g1[3] = (int)((tensor_d1 >> 16) | (tile_d0 << 16));    // tile_dim0
    g1[4] = (int)(tile_d1 & 0xFFFFu);                      // tile_dim1, tile_dim2=0
    g1[5] = (int)stride0;                                  // tensor_dim0_stride lo32
    g1[6] = 0;
    g1[7] = 0;
    v4i z4 = {0, 0, 0, 0};
    v8i z8 = {0, 0, 0, 0, 0, 0, 0, 0};
    __builtin_amdgcn_tensor_load_to_lds(g0, g1, z4, z4, z8, 0);
}

// ---------------------------------------------------------------------------
// 1) Conversions
// ---------------------------------------------------------------------------
__global__ __launch_bounds__(256) void k_cvt_x(const float* __restrict__ x,
                                               unsigned short* __restrict__ xb,
                                               int n) {
    int i = blockIdx.x * 256 + threadIdx.x;
    if (i < n) xb[i] = f2bfu(x[i]);
}

// W [K=1024, N=1024] f32 row-major  ->  Wt [N,K] bf16
__global__ __launch_bounds__(256) void k_cvt_wT(const float* __restrict__ w,
                                                unsigned short* __restrict__ wt) {
    int i = blockIdx.x * 256 + threadIdx.x;      // i < 1024*1024
    int k = i >> 10, n = i & 1023;
    wt[(size_t)n * DMODEL + k] = f2bfu(w[i]);
}

// ---------------------------------------------------------------------------
// 2) QKV projection GEMM + fused RMSNorm epilogue
//    grid (16, 16, 3) block 256: blockIdx.x = head (N tile of 64),
//    each wave: 32 rows (two 16-row M tiles sharing B operands).
// ---------------------------------------------------------------------------
__global__ __launch_bounds__(256) void k_gemm_qkv(
    const unsigned short* __restrict__ xb_,
    const unsigned short* __restrict__ wq_,
    const unsigned short* __restrict__ wk_,
    const unsigned short* __restrict__ wv_,
    const float* __restrict__ qnw,
    const float* __restrict__ knw,
    unsigned short* __restrict__ qn_,
    unsigned short* __restrict__ kn_,
    unsigned short* __restrict__ vt_) {

    const int z = blockIdx.z;
    const __bf16* X  = (const __bf16*)xb_;
    const __bf16* Wt = (const __bf16*)(z == 0 ? wq_ : (z == 1 ? wk_ : wv_));

    const int lane = threadIdx.x & 31;
    const int lh   = lane >> 4;       // lane half
    const int lm   = lane & 15;
    const int wave = threadIdx.x >> 5;
    const int m0   = (blockIdx.y * 8 + wave) * 32;
    const int h    = blockIdx.x;
    const int n0   = h * 64;

    v8f c[2][4];
#pragma unroll
    for (int u = 0; u < 2; ++u)
#pragma unroll
        for (int t = 0; t < 4; ++t)
#pragma unroll
            for (int r = 0; r < 8; ++r) c[u][t][r] = 0.0f;

    const __bf16* arow0 = X + (size_t)(m0 + lm) * DMODEL + 8 * lh;
    const __bf16* arow1 = arow0 + (size_t)16 * DMODEL;
    const __bf16* brow[4];
#pragma unroll
    for (int t = 0; t < 4; ++t)
        brow[t] = Wt + (size_t)(n0 + 16 * t + lm) * DMODEL + 8 * lh;

    for (int k0 = 0; k0 < DMODEL; k0 += 32) {
        v16bf a0 = make_a(arow0 + k0);
        v16bf a1 = make_a(arow1 + k0);
#pragma unroll
        for (int t = 0; t < 4; ++t) {
            v16bf b = make_a(brow[t] + k0);
            c[0][t] = wmma_bf16(a0, b, c[0][t]);
            c[1][t] = wmma_bf16(a1, b, c[1][t]);
        }
    }

    if (z < 2) {
        // fused per-head RMSNorm (+ fold softmax 1/8 scale into Q)
        const float* nw = (z == 0) ? qnw : knw;
        const float  sc = (z == 0) ? 0.125f : 1.0f;
        unsigned short* dst = (z == 0) ? qn_ : kn_;
#pragma unroll
        for (int u = 0; u < 2; ++u)
#pragma unroll
        for (int r = 0; r < 8; ++r) {
            float ss = 0.0f;
#pragma unroll
            for (int t = 0; t < 4; ++t) ss += c[u][t][r] * c[u][t][r];
            ss += __shfl_xor(ss, 1);
            ss += __shfl_xor(ss, 2);
            ss += __shfl_xor(ss, 4);
            ss += __shfl_xor(ss, 8);
            float rn = rsqrtf(ss * (1.0f / 64.0f) + 1e-5f) * sc;
            int m = m0 + 16 * u + r + 8 * lh;
            int bb = m >> 11, tok = m & (SEQ - 1);
            size_t base = (((size_t)bb * NHEADS + h) * SEQ + tok) * HDIM;
#pragma unroll
            for (int t = 0; t < 4; ++t) {
                int d = 16 * t + lm;
                dst[base + d] = f2bfu(c[u][t][r] * rn * nw[d]);
            }
        }
    } else {
        // V: store transposed [B,H,64,T] bf16
#pragma unroll
        for (int u = 0; u < 2; ++u)
#pragma unroll
        for (int r = 0; r < 8; ++r) {
            int m = m0 + 16 * u + r + 8 * lh;
            int bb = m >> 11, tok = m & (SEQ - 1);
#pragma unroll
            for (int t = 0; t < 4; ++t) {
                int d = 16 * t + lm;
                vt_[(((size_t)bb * NHEADS + h) * HDIM + d) * SEQ + tok] =
                    f2bfu(c[u][t][r]);
            }
        }
    }
}

// ---------------------------------------------------------------------------
// 3) Flash attention. grid (B*H, T/128) block 256 (8 waves).
//    All waves share one (b,h); K/V key-blocks (32 keys) are staged into LDS
//    by wave 0 via TDM (double-buffered), then all waves consume via ds loads.
// ---------------------------------------------------------------------------
__global__ __launch_bounds__(256) void k_flash(
    const unsigned short* __restrict__ qn_,
    const unsigned short* __restrict__ kn_,
    const unsigned short* __restrict__ vt_,
    unsigned short* __restrict__ ob_) {

    __shared__ __align__(16) unsigned short sK[2][32 * HDIM];  // 2 x 4 KB
    __shared__ __align__(16) unsigned short sV[2][HDIM * 32];  // 2 x 4 KB

    const int bh = blockIdx.x;
    const int b  = bh >> 4, h = bh & 15;
    const int lane = threadIdx.x & 31;
    const int lh   = lane >> 4;
    const int lm   = lane & 15;
    const int wave = threadIdx.x >> 5;
    const int q0   = blockIdx.y * 128 + wave * 16;

    const __bf16* Qp = (const __bf16*)qn_ + ((size_t)bh * SEQ + q0) * HDIM;
    const unsigned short* Kg = kn_ + (size_t)bh * SEQ * HDIM;
    const unsigned short* Vg = vt_ + (size_t)bh * HDIM * SEQ;

    const int klane = 8 * lh;
    const int aoff  = 8 * lh;

    // Q^T B-operands (reused across all key blocks)
    v16bf bq0 = make_a(Qp + lm * HDIM + klane);
    v16bf bq1 = make_a(Qp + lm * HDIM + 32 + klane);

    float mrun = -1e30f, lrun = 0.0f;
    v8f o[4];
#pragma unroll
    for (int t = 0; t < 4; ++t)
#pragma unroll
        for (int r = 0; r < 8; ++r) o[t][r] = 0.0f;

    // prologue: stage key-block 0
    if (wave == 0) {
        tdm_load_2d((unsigned)(unsigned long long)&sK[0][0], Kg,
                    HDIM, SEQ, HDIM, 32, HDIM);
        tdm_load_2d((unsigned)(unsigned long long)&sV[0][0], Vg,
                    SEQ, HDIM, 32, HDIM, SEQ);
        __builtin_amdgcn_s_wait_tensorcnt(0);
    }
    __syncthreads();

    int buf = 0;
    for (int kb = 0; kb < SEQ; kb += 32) {
        // async-stage next key block while computing this one
        if (wave == 0 && (kb + 32) < SEQ) {
            tdm_load_2d((unsigned)(unsigned long long)&sK[buf ^ 1][0],
                        Kg + (size_t)(kb + 32) * HDIM,
                        HDIM, SEQ, HDIM, 32, HDIM);
            tdm_load_2d((unsigned)(unsigned long long)&sV[buf ^ 1][0],
                        Vg + (kb + 32),
                        SEQ, HDIM, 32, HDIM, SEQ);
        }

        const __bf16* kt = (const __bf16*)&sK[buf][0];
        const __bf16* vt = (const __bf16*)&sV[buf][0];

        // ---- S^T tiles: tile0 keys kb..kb+15, tile1 keys kb+16..kb+31
        v8f s0, s1;
#pragma unroll
        for (int r = 0; r < 8; ++r) { s0[r] = 0.0f; s1[r] = 0.0f; }
        {
            const __bf16* kr = kt + lm * HDIM;
            s0 = wmma_bf16(make_a(kr + aoff),      bq0, s0);
            s0 = wmma_bf16(make_a(kr + 32 + aoff), bq1, s0);
        }
        {
            const __bf16* kr = kt + (16 + lm) * HDIM;
            s1 = wmma_bf16(make_a(kr + aoff),      bq0, s1);
            s1 = wmma_bf16(make_a(kr + 32 + aoff), bq1, s1);
        }

        // ---- online softmax (lane = query domain; pair lanes l, l^16)
        float bm = -1e30f;
#pragma unroll
        for (int r = 0; r < 8; ++r) {
            bm = fmaxf(bm, s0[r]); bm = fmaxf(bm, s1[r]);
        }
        bm = fmaxf(bm, __shfl_xor(bm, 16));
        float mnew  = fmaxf(mrun, bm);
        float alpha = __expf(mrun - mnew);
        mrun = mnew;

        float pa[8], pb[8], rs = 0.0f;
#pragma unroll
        for (int r = 0; r < 8; ++r) {
            pa[r] = __expf(s0[r] - mnew); rs += pa[r];
            pb[r] = __expf(s1[r] - mnew); rs += pb[r];
        }
        rs += __shfl_xor(rs, 16);
        lrun = lrun * alpha + rs;

        // P: C-layout maps directly onto bf16 A-layout (no cross-lane moves)
        v16bf ap;
#pragma unroll
        for (int j = 0; j < 8; ++j) { ap[j] = f2bf(pa[j]); ap[8 + j] = f2bf(pb[j]); }

        // rescale O rows by alpha (broadcast per C-layout row)
        float ab[8];
#pragma unroll
        for (int r = 0; r < 8; ++r) ab[r] = __shfl(alpha, r + 8 * lh);
#pragma unroll
        for (int t = 0; t < 4; ++t)
#pragma unroll
            for (int r = 0; r < 8; ++r) o[t][r] *= ab[r];

        // O += P x V  (V^T tile in LDS -> contiguous B-operand ds loads)
#pragma unroll
        for (int t = 0; t < 4; ++t) {
            v16bf bv = make_a(vt + (16 * t + lm) * 32 + klane);
            o[t] = wmma_bf16(ap, bv, o[t]);
        }

        // publish next buffer: issuing wave waits TDM, then all waves sync
        if (wave == 0) __builtin_amdgcn_s_wait_tensorcnt(0);
        __syncthreads();
        buf ^= 1;
    }

    // normalize by row sum and store bf16 [B, T, H*64]
    float linv = 1.0f / lrun;
    float lb[8];
#pragma unroll
    for (int r = 0; r < 8; ++r) lb[r] = __shfl(linv, r + 8 * lh);
#pragma unroll
    for (int r = 0; r < 8; ++r) {
        int tok = q0 + r + 8 * lh;
        size_t base = ((size_t)b * SEQ + tok) * DMODEL + h * HDIM;
#pragma unroll
        for (int t = 0; t < 4; ++t)
            ob_[base + 16 * t + lm] = f2bfu(o[t][r] * lb[r]);
    }
}

// ---------------------------------------------------------------------------
// 4) Output projection GEMM -> f32 d_out. grid (16, 16) block 256,
//    two 16-row M tiles per wave.
// ---------------------------------------------------------------------------
__global__ __launch_bounds__(256) void k_gemm_out(
    const unsigned short* __restrict__ ob_,
    const unsigned short* __restrict__ wot_,
    float* __restrict__ out) {

    const __bf16* X  = (const __bf16*)ob_;
    const __bf16* Wt = (const __bf16*)wot_;
    const int lane = threadIdx.x & 31;
    const int lh   = lane >> 4;
    const int lm   = lane & 15;
    const int wave = threadIdx.x >> 5;
    const int m0   = (blockIdx.y * 8 + wave) * 32;
    const int n0   = blockIdx.x * 64;

    v8f c[2][4];
#pragma unroll
    for (int u = 0; u < 2; ++u)
#pragma unroll
        for (int t = 0; t < 4; ++t)
#pragma unroll
            for (int r = 0; r < 8; ++r) c[u][t][r] = 0.0f;

    const __bf16* arow0 = X + (size_t)(m0 + lm) * DMODEL + 8 * lh;
    const __bf16* arow1 = arow0 + (size_t)16 * DMODEL;
    const __bf16* brow[4];
#pragma unroll
    for (int t = 0; t < 4; ++t)
        brow[t] = Wt + (size_t)(n0 + 16 * t + lm) * DMODEL + 8 * lh;

    for (int k0 = 0; k0 < DMODEL; k0 += 32) {
        v16bf a0 = make_a(arow0 + k0);
        v16bf a1 = make_a(arow1 + k0);
#pragma unroll
        for (int t = 0; t < 4; ++t) {
            v16bf b = make_a(brow[t] + k0);
            c[0][t] = wmma_bf16(a0, b, c[0][t]);
            c[1][t] = wmma_bf16(a1, b, c[1][t]);
        }
    }

#pragma unroll
    for (int u = 0; u < 2; ++u)
#pragma unroll
    for (int r = 0; r < 8; ++r) {
        int m = m0 + 16 * u + r + 8 * lh;
#pragma unroll
        for (int t = 0; t < 4; ++t)
            out[(size_t)m * DMODEL + n0 + 16 * t + lm] = c[u][t][r];
    }
}

// ---------------------------------------------------------------------------
// Host launcher
// ---------------------------------------------------------------------------
extern "C" void kernel_launch(void* const* d_in, const int* in_sizes, int n_in,
                              void* d_out, int out_size, void* d_ws, size_t ws_size,
                              hipStream_t stream) {
    const float* x   = (const float*)d_in[0];
    const float* wq  = (const float*)d_in[1];
    const float* wk  = (const float*)d_in[2];
    const float* wv  = (const float*)d_in[3];
    const float* wo  = (const float*)d_in[4];
    const float* qnw = (const float*)d_in[5];
    const float* knw = (const float*)d_in[6];

    char* ws = (char*)d_ws;
    const size_t MB = 1024ull * 1024ull;
    unsigned short* xb  = (unsigned short*)(ws + 0 * MB);   // 8 MB  [4096,1024] bf16
    unsigned short* wqt = (unsigned short*)(ws + 8 * MB);   // 2 MB  [N,K] bf16
    unsigned short* wkt = (unsigned short*)(ws + 10 * MB);
    unsigned short* wvt = (unsigned short*)(ws + 12 * MB);
    unsigned short* wot = (unsigned short*)(ws + 14 * MB);
    unsigned short* qn  = (unsigned short*)(ws + 16 * MB);  // 8 MB [B,H,T,64]
    unsigned short* kn  = (unsigned short*)(ws + 24 * MB);  // 8 MB [B,H,T,64]
    unsigned short* vt  = (unsigned short*)(ws + 32 * MB);  // 8 MB [B,H,64,T]
    unsigned short* ob  = (unsigned short*)(ws + 40 * MB);  // 8 MB [B,T,1024]

    const int nx = MROWS * DMODEL;                 // 4M
    k_cvt_x<<<nx / 256, 256, 0, stream>>>(x, xb, nx);
    k_cvt_wT<<<(DMODEL * DMODEL) / 256, 256, 0, stream>>>(wq, wqt);
    k_cvt_wT<<<(DMODEL * DMODEL) / 256, 256, 0, stream>>>(wk, wkt);
    k_cvt_wT<<<(DMODEL * DMODEL) / 256, 256, 0, stream>>>(wv, wvt);
    k_cvt_wT<<<(DMODEL * DMODEL) / 256, 256, 0, stream>>>(wo, wot);

    k_gemm_qkv<<<dim3(NHEADS, MROWS / 256, 3), 256, 0, stream>>>(
        xb, wqt, wkt, wvt, qnw, knw, qn, kn, vt);

    k_flash<<<dim3(BATCH * NHEADS, SEQ / 128), 256, 0, stream>>>(qn, kn, vt, ob);

    k_gemm_out<<<dim3(NHEADS, MROWS / 256), 256, 0, stream>>>(ob, wot,
                                                              (float*)d_out);
}
